// MultiBoxLoss_30700426232512
// MI455X (gfx1250) — matrix operand order repaired and checked by reading, not compiled, not run on hardware
//
#include <hip/hip_runtime.h>
#include <cstdint>

#define B 8
#define P 19248
#define O 16
#define C 81
#define M2 256
#define TILE_R 128
#define MWAVES 4

typedef __attribute__((ext_vector_type(16))) _Float16 v16h;
typedef __attribute__((ext_vector_type(8)))  float    v8f;
typedef __attribute__((ext_vector_type(4)))  unsigned u32x4;
typedef __attribute__((ext_vector_type(8)))  int      i32x8;
typedef __attribute__((ext_vector_type(4)))  int      i32x4;

// ---------------------------------------------------------------- init
__global__ void k_init(unsigned long long* objkey, int* cnt, float* acc) {
  int i = threadIdx.x + blockIdx.x * blockDim.x;
  if (i < B * O) objkey[i] = 0ull;
  if (i < B) cnt[i] = 0;
  if (i < 4) acc[i] = 0.f;
}

// ------------------------------------------------- IoU match (per prior)
__global__ void k_overlap(const float* __restrict__ priors,
                          const float* __restrict__ gtb,
                          float* __restrict__ bto, int* __restrict__ bti,
                          unsigned long long* __restrict__ objkey) {
  int b = blockIdx.y;
  int p = blockIdx.x * blockDim.x + threadIdx.x;
  __shared__ float4 tb[O];
  __shared__ float ta[O];
  __shared__ unsigned long long skey[O];
  if (threadIdx.x < O) {
    const float* g = gtb + ((size_t)b * O + threadIdx.x) * 4;
    float4 t = {g[0], g[1], g[2], g[3]};
    tb[threadIdx.x] = t;
    ta[threadIdx.x] = (t.z - t.x) * (t.w - t.y);
    skey[threadIdx.x] = 0ull;
  }
  __syncthreads();
  if (p < P) {
    const float* pr = priors + (size_t)p * 4;
    float w = pr[2], h = pr[3];
    float x1 = pr[0] - 0.5f * w, y1 = pr[1] - 0.5f * h;
    float x2 = pr[0] + 0.5f * w, y2 = pr[1] + 0.5f * h;
    float ap = w * h;
    float best = -1.f; int bj = 0;
    #pragma unroll
    for (int j = 0; j < O; ++j) {
      float4 t = tb[j];
      float iw = fmaxf(fminf(t.z, x2) - fmaxf(t.x, x1), 0.f);
      float ih = fmaxf(fminf(t.w, y2) - fmaxf(t.y, y1), 0.f);
      float inter = iw * ih;
      float iou = inter / (ta[j] + ap - inter);
      if (iou > best) { best = iou; bj = j; }   // first-max like jnp.argmax
      unsigned long long key =
          ((unsigned long long)__float_as_uint(iou) << 32) |
          (unsigned long long)(0xFFFFFFFFu - (unsigned)p);  // tie -> lower p
      atomicMax(&skey[j], key);
    }
    bto[(size_t)b * P + p] = best;
    bti[(size_t)b * P + p] = bj;
  }
  __syncthreads();
  if (threadIdx.x < O) atomicMax(&objkey[b * O + threadIdx.x], skey[threadIdx.x]);
}

// -------------------------------------- forced-match fixup (sequential j)
__global__ void k_fixup(const unsigned long long* __restrict__ objkey,
                        float* __restrict__ bto, int* __restrict__ bti) {
  int b = threadIdx.x;
  if (b < B) {
    for (int j = 0; j < O; ++j) {   // ascending j: last-wins == jax loop
      unsigned long long k = objkey[b * O + j];
      int p = (int)(0xFFFFFFFFu - (unsigned)(k & 0xFFFFFFFFull));
      bto[(size_t)b * P + p] = 2.0f;
      bti[(size_t)b * P + p] = j;
    }
  }
}

// -------------------- conf_t, positive compaction, smooth-L1 loc loss
__global__ void k_assign(const float* __restrict__ loc,
                         const float* __restrict__ priors,
                         const float* __restrict__ gtb,
                         const int* __restrict__ gtl,
                         const float* __restrict__ bto,
                         const int* __restrict__ bti,
                         int* __restrict__ conf_t, int* __restrict__ poslist,
                         int* __restrict__ cnt, float* __restrict__ acc) {
  int b = blockIdx.y;
  int p = blockIdx.x * blockDim.x + threadIdx.x;
  if (p >= P) return;
  size_t i = (size_t)b * P + p;
  float ov = bto[i];
  int idx = bti[i];
  int c = (ov < 0.5f) ? 0 : (gtl[b * O + idx] + 1);
  conf_t[i] = c;
  if (c > 0) {
    int slot = atomicAdd(&cnt[b], 1);
    poslist[(size_t)b * P + slot] = p;
    const float* t = gtb + ((size_t)b * O + idx) * 4;
    const float* pr = priors + (size_t)p * 4;
    float g0 = ((t[0] + t[2]) * 0.5f - pr[0]) / (0.1f * pr[2]);
    float g1 = ((t[1] + t[3]) * 0.5f - pr[1]) / (0.1f * pr[3]);
    float g2 = __logf((t[2] - t[0]) / pr[2]) / 0.2f;
    float g3 = __logf((t[3] - t[1]) / pr[3]) / 0.2f;
    float gt4[4] = {g0, g1, g2, g3};
    const float* ld = loc + i * 4;
    float s = 0.f;
    #pragma unroll
    for (int k = 0; k < 4; ++k) {
      float d = ld[k] - gt4[k], ad = fabsf(d);
      s += (ad < 1.f) ? 0.5f * d * d : (ad - 0.5f);
    }
    atomicAdd(&acc[0], s);
  }
}

// ---------------- TDM-staged logsumexp / CE (bandwidth-dominant kernel)
__device__ __forceinline__ void tdm_load_rows(unsigned ldsoff, const void* gptr,
                                              unsigned rows_total) {
#if defined(__has_builtin) && __has_builtin(__builtin_amdgcn_tensor_load_to_lds)
  unsigned long long ga = (unsigned long long)(uintptr_t)gptr;
  // D# group0: count=1 | lds_addr | global_addr(57b) | type=2
  u32x4 g0 = {1u, ldsoff, (unsigned)ga,
              (unsigned)((ga >> 32) & 0x1FFFFFFull) | (2u << 30)};
  // D# group1: data_size=4B; tensor_dim0=81, tensor_dim1=rows_total,
  //            tile_dim0=81, tile_dim1=TILE_R, dim0_stride=81
  unsigned d0 = 2u << 16;
  unsigned d1 = ((unsigned)C & 0xFFFFu) << 16;
  unsigned d2 = (((unsigned)C >> 16) & 0xFFFFu) | ((rows_total & 0xFFFFu) << 16);
  unsigned d3 = ((rows_total >> 16) & 0xFFFFu) | (((unsigned)C & 0xFFFFu) << 16);
  unsigned d4 = (unsigned)TILE_R & 0xFFFFu;
  unsigned d5 = (unsigned)C;
  i32x8 g1 = {(int)d0, (int)d1, (int)d2, (int)d3, (int)d4, (int)d5, 0, 0};
  i32x4 z4 = {};
#if __clang_major__ >= 23
  i32x8 z8 = {};
  __builtin_amdgcn_tensor_load_to_lds(g0, g1, z4, z4, z8, 0);
#else
  __builtin_amdgcn_tensor_load_to_lds(g0, g1, z4, z4, 0);
#endif
#else
  (void)ldsoff; (void)gptr; (void)rows_total;
#endif
}

__global__ void k_ce(const float* __restrict__ conf,
                     const int* __restrict__ conf_t,
                     float* __restrict__ mine, float* __restrict__ acc) {
  __shared__ float tile[TILE_R * C];
  long long row0 = (long long)blockIdx.x * TILE_R;
#if defined(__has_builtin) && __has_builtin(__builtin_amdgcn_tensor_load_to_lds)
  if (threadIdx.x == 0) {
    tdm_load_rows((unsigned)(uintptr_t)(void*)&tile[0], conf + row0 * C,
                  (unsigned)((long long)B * P));
  }
  __builtin_amdgcn_s_wait_tensorcnt(0);
#else
  {  // fallback staging (keeps kernel compiling on toolchains without TDM builtin)
    long long r = row0 + threadIdx.x;
    if (r < (long long)B * P)
      for (int c2 = 0; c2 < C; ++c2) tile[threadIdx.x * C + c2] = conf[r * C + c2];
  }
#endif
  __syncthreads();
  long long row = row0 + threadIdx.x;
  if (row < (long long)B * P) {
    const float* v = &tile[threadIdx.x * C];
    float mx = -1e30f;
    for (int c2 = 0; c2 < C; ++c2) mx = fmaxf(mx, v[c2]);
    float s = 0.f;
    for (int c2 = 0; c2 < C; ++c2) s += __expf(v[c2] - mx);
    int gc = conf_t[row];
    float ce = mx + __logf(s) - v[gc];
    bool pos = gc > 0;
    mine[row] = pos ? 0.f : ce;
    if (pos) atomicAdd(&acc[1], ce);   // positives always selected
  }
}

// --------- hard-negative mining: sum of top-k via threshold bisection
__global__ void k_topk(const float* __restrict__ mine,
                       const int* __restrict__ cnt, float* __restrict__ acc) {
  int b = blockIdx.x;
  int tid = threadIdx.x;
  __shared__ float rf[256];
  __shared__ int ri[256];
  int k = min(3 * cnt[b], P - 1);
  const float* mb = mine + (size_t)b * P;
  float mx = 0.f;
  for (int i = tid; i < P; i += 256) mx = fmaxf(mx, mb[i]);
  rf[tid] = mx;
  __syncthreads();
  for (int s = 128; s > 0; s >>= 1) {
    if (tid < s) rf[tid] = fmaxf(rf[tid], rf[tid + s]);
    __syncthreads();
  }
  float lo = 0.f, hi = rf[0] + 1e-6f;
  __syncthreads();
  if (k <= 0) return;
  for (int it = 0; it < 40; ++it) {
    float mid = 0.5f * (lo + hi);
    int c = 0;
    for (int i = tid; i < P; i += 256) c += (mb[i] > mid) ? 1 : 0;
    ri[tid] = c;
    __syncthreads();
    for (int s = 128; s > 0; s >>= 1) {
      if (tid < s) ri[tid] += ri[tid + s];
      __syncthreads();
    }
    int tot = ri[0];
    __syncthreads();
    if (tot >= k) lo = mid; else hi = mid;
  }
  float sv = 0.f; int cg = 0;
  for (int i = tid; i < P; i += 256) {
    float x = mb[i];
    if (x > lo) { sv += x; cg++; }
  }
  rf[tid] = sv; ri[tid] = cg;
  __syncthreads();
  for (int s = 128; s > 0; s >>= 1) {
    if (tid < s) { rf[tid] += rf[tid + s]; ri[tid] += ri[tid + s]; }
    __syncthreads();
  }
  if (tid == 0) atomicAdd(&acc[1], rf[0] + (float)(k - ri[0]) * lo);
}

// ------------------- mask BCE over positives via WMMA f16 (16 priors/tile)
__global__ void k_mask(const float* __restrict__ mask_data,
                       const float* __restrict__ gtm,
                       const int* __restrict__ poslist,
                       const int* __restrict__ bti,
                       const int* __restrict__ cnt, float* __restrict__ acc) {
  int b = blockIdx.y;
  int lane = threadIdx.x & 31;
  int wid = threadIdx.x >> 5;
  int tile = blockIdx.x * MWAVES + wid;
  int n = cnt[b];
  if (tile >= ((n + 15) >> 4)) return;     // wave-uniform exit
  int h = lane >> 4, sub = lane & 15;

  // B fragments: 0/1 gt masks, object = sub. B in contiguous k order (same as A): K = 16*h + e per chunk.
  v16h Bf[8];
  const float* gmr = gtm + ((size_t)b * O + sub) * M2 + 16 * h;
  #pragma unroll
  for (int c2 = 0; c2 < 8; ++c2) {
    #pragma unroll
    for (int q = 0; q < 4; ++q) {
      float4 m4 = *(const float4*)(gmr + 32 * c2 + 4 * q);
      Bf[c2][4 * q + 0] = (_Float16)m4.x;
      Bf[c2][4 * q + 1] = (_Float16)m4.y;
      Bf[c2][4 * q + 2] = (_Float16)m4.z;
      Bf[c2][4 * q + 3] = (_Float16)m4.w;
    }
  }

  int li = 16 * tile + sub;
  bool valid = li < n;
  int p = valid ? poslist[(size_t)b * P + li] : 0;   // lane reads safe row 0 if pad
  int myg = valid ? bti[(size_t)b * P + p] : -1;
  float vm = valid ? 1.f : 0.f;                      // branchless zero-mask
  const float* md = mask_data + ((size_t)b * P + p) * M2;

  float bias = 0.f;
  v8f accv = {};
  #pragma unroll
  for (int c2 = 0; c2 < 8; ++c2) {
    v16h Af;
    // A fragment (contiguous, matches B): VGPR v -> K=2v,2v+1
    // +16 per lane-half h. Pairs are K-consecutive -> float2 loads, no branches.
    #pragma unroll
    for (int v = 0; v < 8; ++v) {
      int base = 2 * v + 16 * h;
      float2 x2 = *(const float2*)(md + 32 * c2 + base);
      float x0 = fminf(fmaxf(x2.x, 1e-7f), 1.f - 1e-7f);
      float x1 = fminf(fmaxf(x2.y, 1e-7f), 1.f - 1e-7f);
      float l1p0 = __logf(1.f - x0);
      float l1p1 = __logf(1.f - x1);
      float w0 = (l1p0 - __logf(x0)) * vm;
      float w1 = (l1p1 - __logf(x1)) * vm;
      bias -= (l1p0 + l1p1) * vm;          // each tile element touched exactly once
      Af[2 * v + 0] = (_Float16)w0;
      Af[2 * v + 1] = (_Float16)w1;
    }
    accv = __builtin_amdgcn_wmma_f32_16x16x32_f16(
        false, Af, false, Bf[c2], (short)0, accv, false, false);
  }

  float local = bias;
  #pragma unroll
  for (int r = 0; r < 8; ++r) {            // D: lane holds (M = r+8h, N = sub)
    int M = r + 8 * h;
    int g = __shfl(myg, M, 32);
    if (sub == g) local += accv[r];
  }
  for (int s = 16; s > 0; s >>= 1) local += __shfl_xor(local, s, 32);
  if (lane == 0) atomicAdd(&acc[2], local);
}

// ---------------------------------------------------------------- finalize
__global__ void k_final(const int* __restrict__ cnt,
                        const float* __restrict__ acc, float* __restrict__ out) {
  if (threadIdx.x == 0) {
    int n = 0;
    for (int b = 0; b < B; ++b) n += cnt[b];
    float N = (float)n;
    out[0] = acc[0] / N;
    out[1] = acc[1] / N;
    out[2] = (acc[2] / (N * (float)M2) * 100.f) / N;
  }
}

extern "C" void kernel_launch(void* const* d_in, const int* in_sizes, int n_in,
                              void* d_out, int out_size, void* d_ws, size_t ws_size,
                              hipStream_t stream) {
  const float* loc    = (const float*)d_in[0];
  const float* conf   = (const float*)d_in[1];
  const float* maskd  = (const float*)d_in[2];
  const float* priors = (const float*)d_in[3];
  const float* gtb    = (const float*)d_in[4];
  const int*   gtl    = (const int*)d_in[5];
  const float* gtm    = (const float*)d_in[6];
  float* out = (float*)d_out;

  char* w = (char*)d_ws;
  auto alloc = [&](size_t bytes) {
    char* r = w;
    w += (bytes + 255) & ~(size_t)255;
    return r;
  };
  float* bto = (float*)alloc(sizeof(float) * B * P);
  int* bti = (int*)alloc(sizeof(int) * B * P);
  int* conf_t = (int*)alloc(sizeof(int) * B * P);
  float* mine = (float*)alloc(sizeof(float) * B * P);
  int* poslist = (int*)alloc(sizeof(int) * B * P);
  unsigned long long* objkey =
      (unsigned long long*)alloc(sizeof(unsigned long long) * B * O);
  int* cnt = (int*)alloc(sizeof(int) * B);
  float* acc = (float*)alloc(sizeof(float) * 4);

  k_init<<<1, 256, 0, stream>>>(objkey, cnt, acc);
  dim3 gpp((P + 255) / 256, B);
  k_overlap<<<gpp, 256, 0, stream>>>(priors, gtb, bto, bti, objkey);
  k_fixup<<<1, B, 0, stream>>>(objkey, bto, bti);
  k_assign<<<gpp, 256, 0, stream>>>(loc, priors, gtb, gtl, bto, bti,
                                    conf_t, poslist, cnt, acc);
  int nrows = B * P;
  k_ce<<<(nrows + TILE_R - 1) / TILE_R, TILE_R, 0, stream>>>(conf, conf_t, mine, acc);
  k_topk<<<B, 256, 0, stream>>>(mine, cnt, acc);
  dim3 gmk((((P + 15) / 16) + MWAVES - 1) / MWAVES, B);
  k_mask<<<gmk, 32 * MWAVES, 0, stream>>>(maskd, gtm, poslist, bti, cnt, acc);
  k_final<<<1, 1, 0, stream>>>(cnt, acc, out);
}
